// MultiHeadAttention_53807350284345
// MI455X (gfx1250) — matrix operand order repaired and checked by reading, not compiled, not run on hardware
//
#include <hip/hip_runtime.h>
#include <hip/hip_bf16.h>

// ---------------------------------------------------------------------------
// CDNA5 (gfx1250) fused MHA: QKV proj + RoPE -> flash attention -> out proj
// All matmuls on v_wmma_f32_16x16x32_bf16 (f32 accumulate). wave32.
// ---------------------------------------------------------------------------

typedef __attribute__((ext_vector_type(8)))  __bf16 v8bf;
typedef __attribute__((ext_vector_type(16))) __bf16 v16bf;
typedef __attribute__((ext_vector_type(8)))  float  v8f;

__device__ __forceinline__ __bf16 f2bf(float f) {
    union { float f; unsigned u; } a; a.f = f;
    unsigned r = a.u + 0x7FFFu + ((a.u >> 16) & 1u);   // round-to-nearest-even
    union { unsigned short s; __bf16 b; } o; o.s = (unsigned short)(r >> 16);
    return o.b;
}

// A-matrix fragment (16x32 bf16, MxK), row-major source, row stride ld.
// ISA layout: lane l holds row m=l&15; K = h*8..h*8+7 in elems 0..7 and
// K = 16+h*8..+7 in elems 8..15, h = l>>4.  Two b128 loads.
__device__ __forceinline__ v16bf load_a(const __bf16* base, int ld) {
    const int l = threadIdx.x & 31;
    const __bf16* p = base + (l & 15) * ld + (l >> 4) * 8;
    union { v16bf v; struct { v8bf lo, hi; } h; } u;
    u.h.lo = *(const v8bf*)p;
    u.h.hi = *(const v8bf*)(p + 16);
    return u.v;
}

// B-matrix fragment (32x16 bf16, KxN) loaded from B^T stored row-major
// (row = n, col = k, stride ld).  ISA layout: lane l holds column n=l&15;
// K = h*8..h*8+7 in elems 0..7 and K = 16+h*8..+7 in elems 8..15, h = l>>4.  Two b128 loads.
__device__ __forceinline__ v16bf load_b(const __bf16* base, int ld) {
    const int l = threadIdx.x & 31;
    const __bf16* p = base + (l & 15) * ld + (l >> 4) * 8;
    union { v16bf v; struct { v8bf lo, hi; } h; } u;
    u.h.lo = *(const v8bf*)p;
    u.h.hi = *(const v8bf*)(p + 16);
    return u.v;
}

__device__ __forceinline__ v8f wmma_bf16(v16bf a, v16bf b, v8f c) {
    // (neg_a, A, neg_b, B, c_mod, C, reuse_a, reuse_b)
    return __builtin_amdgcn_wmma_f32_16x16x32_bf16(false, a, false, b,
                                                   (short)0, c, false, false);
}

// ---------------------------------------------------------------------------
// f32 -> bf16 bulk convert (4 elems / thread, b64 stores)
// ---------------------------------------------------------------------------
__global__ void cvt_bf16_kernel(const float* __restrict__ in,
                                __bf16* __restrict__ out, int n) {
    int i = (blockIdx.x * blockDim.x + threadIdx.x) * 4;
    if (i + 3 < n) {
        float4 v = *(const float4*)(in + i);
        union { __bf16 b[4]; unsigned long long u; } o;
        o.b[0] = f2bf(v.x); o.b[1] = f2bf(v.y);
        o.b[2] = f2bf(v.z); o.b[3] = f2bf(v.w);
        *(unsigned long long*)(out + i) = o.u;
    }
}

// ---------------------------------------------------------------------------
// NT GEMM:  C[M,N] = A[M,K] * W[N,K]^T   (both bf16 row-major, K contiguous)
// Wave tile: 32(M) x 64(N), K-step 32, 8 WMMA per step. Block = 8 waves
// covering 256 rows. Relies on 192MB L2 for A/W tile reuse (no LDS staging).
// mode 0: f32 out (Cf)          mode 1: bf16 out + RoPE (Cb, pos)
// mode 2: bf16 out (Cb)         mode 3: bf16 transposed out (Ct, [N][M])
// ---------------------------------------------------------------------------
__global__ void __launch_bounds__(256)
gemm_nt_kernel(const __bf16* __restrict__ A, const __bf16* __restrict__ W,
               float* __restrict__ Cf, __bf16* __restrict__ Cb,
               __bf16* __restrict__ Ct, const int* __restrict__ pos,
               int M, int N, int K, int mode) {
    const int wave = threadIdx.x >> 5;
    const int lane = threadIdx.x & 31;
    const int m0 = blockIdx.x * 256 + wave * 32;
    const int n0 = blockIdx.y * 64;

    v8f acc[2][4] = {};
    #pragma unroll 2
    for (int k0 = 0; k0 < K; k0 += 32) {
        v16bf a0 = load_a(A + (size_t)m0 * K + k0, K);
        v16bf a1 = load_a(A + (size_t)(m0 + 16) * K + k0, K);
        #pragma unroll
        for (int t = 0; t < 4; t++) {
            v16bf b = load_b(W + (size_t)(n0 + t * 16) * K + k0, K);
            acc[0][t] = wmma_bf16(a0, b, acc[0][t]);
            acc[1][t] = wmma_bf16(a1, b, acc[1][t]);
        }
    }

    const int n  = lane & 15;
    const int mb = (lane >> 4) * 8;

    if (mode == 1) {
        // Fused RoPE: head dim = 64 and n0 is 64-aligned -> this wave's N-tile
        // is exactly one head. d = t*16+n, pair (2i,2i+1) via shfl_xor(1).
        #pragma unroll
        for (int mi = 0; mi < 2; mi++)
        #pragma unroll
        for (int t = 0; t < 4; t++) {
            int d = t * 16 + n;
            float invf = __powf(100000.0f, -(float)(d >> 1) * (1.0f / 32.0f));
            bool odd = (d & 1) != 0;
            #pragma unroll
            for (int r = 0; r < 8; r++) {
                float v = acc[mi][t][r];
                float partner = __shfl_xor(v, 1);
                int row = m0 + mi * 16 + mb + r;
                float ang = (float)pos[row] * invf;
                float sn = sinf(ang), cs = cosf(ang);
                acc[mi][t][r] = odd ? (partner * sn + v * cs)
                                    : (v * cs - partner * sn);
            }
        }
    }

    if (mode == 3) {
        // store transposed: Ct[e][s], 8 contiguous rows per lane -> b128 store
        #pragma unroll
        for (int mi = 0; mi < 2; mi++)
        #pragma unroll
        for (int t = 0; t < 4; t++) {
            int e = n0 + t * 16 + n;
            union { v8bf v; __bf16 b[8]; } o;
            #pragma unroll
            for (int r = 0; r < 8; r++) o.b[r] = f2bf(acc[mi][t][r]);
            *(v8bf*)(Ct + (size_t)e * M + m0 + mi * 16 + mb) = o.v;
        }
    } else if (mode == 0) {
        #pragma unroll
        for (int mi = 0; mi < 2; mi++)
        #pragma unroll
        for (int t = 0; t < 4; t++)
        #pragma unroll
        for (int r = 0; r < 8; r++)
            Cf[(size_t)(m0 + mi * 16 + mb + r) * N + n0 + t * 16 + n] =
                acc[mi][t][r];
    } else {
        #pragma unroll
        for (int mi = 0; mi < 2; mi++)
        #pragma unroll
        for (int t = 0; t < 4; t++)
        #pragma unroll
        for (int r = 0; r < 8; r++)
            Cb[(size_t)(m0 + mi * 16 + mb + r) * N + n0 + t * 16 + n] =
                f2bf(acc[mi][t][r]);
    }
}

// ---------------------------------------------------------------------------
// Causal flash attention, one wave per (head, 16-query tile).
// S=4096, H=16, d_k=64, scale = 1/8.  Online softmax in f32.
// P (score->A-matrix) layout bounce goes through per-wave LDS (1KB/wave).
// ---------------------------------------------------------------------------
__global__ void __launch_bounds__(256)
flash_attn_kernel(const __bf16* __restrict__ Q, const __bf16* __restrict__ Kb,
                  const __bf16* __restrict__ Vt, __bf16* __restrict__ O) {
    __shared__ __align__(16) __bf16 plds[8][16 * 32];
    const int lane = threadIdx.x & 31, wave = threadIdx.x >> 5;
    const int id = blockIdx.x * 8 + wave;
    const int h = id & 15;
    const int q0 = (id >> 4) * 16;
    const int n = lane & 15, mb = (lane >> 4) * 8;

    const __bf16* Qh = Q + (size_t)q0 * 1024 + h * 64;
    const v16bf qa0 = load_a(Qh, 1024);
    const v16bf qa1 = load_a(Qh + 32, 1024);

    v8f o0 = {}, o1 = {}, o2 = {}, o3 = {};
    float mrow[8], lrow[8];
    #pragma unroll
    for (int r = 0; r < 8; r++) { mrow[r] = -3.0e38f; lrow[r] = 0.0f; }

    for (int k0 = 0; k0 < q0 + 16; k0 += 32) {
        // S tile = Q (16x64) * K^T (64x32): two 16-key N-subtiles
        const __bf16* Kp = Kb + (size_t)k0 * 1024 + h * 64;
        v8f sv0 = {}, sv1 = {};
        sv0 = wmma_bf16(qa0, load_b(Kp, 1024), sv0);
        sv0 = wmma_bf16(qa1, load_b(Kp + 32, 1024), sv0);
        Kp += 16 * 1024;
        sv1 = wmma_bf16(qa0, load_b(Kp, 1024), sv1);
        sv1 = wmma_bf16(qa1, load_b(Kp + 32, 1024), sv1);

        #pragma unroll
        for (int r = 0; r < 8; r++) {
            const int qg = q0 + mb + r;
            float a0 = sv0[r] * 0.125f; if (k0 + n      > qg) a0 = -3.0e38f;
            float a1 = sv1[r] * 0.125f; if (k0 + 16 + n > qg) a1 = -3.0e38f;
            // row max across the 16-lane group (masks 1..8 stay in-group)
            float mx = fmaxf(a0, a1);
            #pragma unroll
            for (int off = 1; off < 16; off <<= 1)
                mx = fmaxf(mx, __shfl_xor(mx, off));
            const float mnew = fmaxf(mrow[r], mx);
            const float alpha = __expf(mrow[r] - mnew);
            const float p0 = __expf(a0 - mnew);
            const float p1 = __expf(a1 - mnew);
            plds[wave][(mb + r) * 32 + n]      = f2bf(p0);
            plds[wave][(mb + r) * 32 + 16 + n] = f2bf(p1);
            float ps = p0 + p1;
            #pragma unroll
            for (int off = 1; off < 16; off <<= 1)
                ps += __shfl_xor(ps, off);
            lrow[r] = lrow[r] * alpha + ps;
            mrow[r] = mnew;
            o0[r] *= alpha; o1[r] *= alpha; o2[r] *= alpha; o3[r] *= alpha;
        }

        // P (16x32) @ V (32x64); same-wave LDS ops are in-order (DScnt)
        const v16bf pa = load_a(&plds[wave][0], 32);
        const __bf16* Vh = Vt + (size_t)(h * 64) * 4096 + k0;
        o0 = wmma_bf16(pa, load_b(Vh + (size_t)0  * 4096, 4096), o0);
        o1 = wmma_bf16(pa, load_b(Vh + (size_t)16 * 4096, 4096), o1);
        o2 = wmma_bf16(pa, load_b(Vh + (size_t)32 * 4096, 4096), o2);
        o3 = wmma_bf16(pa, load_b(Vh + (size_t)48 * 4096, 4096), o3);
    }

    #pragma unroll
    for (int r = 0; r < 8; r++) {
        const float inv = 1.0f / lrow[r];
        __bf16* Op = O + (size_t)(q0 + mb + r) * 1024 + h * 64;
        Op[0 * 16 + n] = f2bf(o0[r] * inv);
        Op[1 * 16 + n] = f2bf(o1[r] * inv);
        Op[2 * 16 + n] = f2bf(o2[r] * inv);
        Op[3 * 16 + n] = f2bf(o3[r] * inv);
    }
}

// ---------------------------------------------------------------------------
extern "C" void kernel_launch(void* const* d_in, const int* in_sizes, int n_in,
                              void* d_out, int out_size, void* d_ws,
                              size_t ws_size, hipStream_t stream) {
    const float* x  = (const float*)d_in[0];   // [4096,1024]
    const float* Wq = (const float*)d_in[1];   // [1024,1024]
    const float* Wk = (const float*)d_in[2];
    const float* Wv = (const float*)d_in[3];
    const float* Wo = (const float*)d_in[4];
    const int*  pos = (const int*)d_in[5];     // [4096]
    float* out = (float*)d_out;

    const int S = 4096, D = 1024;
    char* w = (char*)d_ws;
    __bf16* xb  = (__bf16*)w; w += (size_t)S * D * 2;
    __bf16* wqb = (__bf16*)w; w += (size_t)D * D * 2;
    __bf16* wkb = (__bf16*)w; w += (size_t)D * D * 2;
    __bf16* wvb = (__bf16*)w; w += (size_t)D * D * 2;
    __bf16* wob = (__bf16*)w; w += (size_t)D * D * 2;
    __bf16* qb  = (__bf16*)w; w += (size_t)S * D * 2;
    __bf16* kb  = (__bf16*)w; w += (size_t)S * D * 2;
    __bf16* vt  = (__bf16*)w; w += (size_t)S * D * 2;   // transposed [D][S]
    __bf16* ab  = (__bf16*)w; w += (size_t)S * D * 2;

    // 1) bf16 conversions
    cvt_bf16_kernel<<<(S * D) / 1024, 256, 0, stream>>>(x,  xb,  S * D);
    cvt_bf16_kernel<<<(D * D) / 1024, 256, 0, stream>>>(Wq, wqb, D * D);
    cvt_bf16_kernel<<<(D * D) / 1024, 256, 0, stream>>>(Wk, wkb, D * D);
    cvt_bf16_kernel<<<(D * D) / 1024, 256, 0, stream>>>(Wv, wvb, D * D);
    cvt_bf16_kernel<<<(D * D) / 1024, 256, 0, stream>>>(Wo, wob, D * D);

    // 2) Q/K projections with fused RoPE; V projection stored transposed
    dim3 g(S / 256, D / 64);
    gemm_nt_kernel<<<g, 256, 0, stream>>>(xb, wqb, nullptr, qb, nullptr, pos,
                                          S, D, D, 1);
    gemm_nt_kernel<<<g, 256, 0, stream>>>(xb, wkb, nullptr, kb, nullptr, pos,
                                          S, D, D, 1);
    gemm_nt_kernel<<<g, 256, 0, stream>>>(xb, wvb, nullptr, nullptr, vt,
                                          nullptr, S, D, D, 3);

    // 3) causal flash attention: 16 heads * 256 q-tiles = 4096 waves
    flash_attn_kernel<<<512, 256, 0, stream>>>(qb, kb, vt, ab);

    // 4) output projection, f32 result
    gemm_nt_kernel<<<g, 256, 0, stream>>>(ab, wob, out, nullptr, nullptr,
                                          nullptr, S, D, D, 0);
}
